// GradMimicryNetwork_47751446396959
// MI455X (gfx1250) — matrix operand order repaired and checked by reading, not compiled, not run on hardware
//
#include <hip/hip_runtime.h>
#include <cstdint>

// ---------------------------------------------------------------------------
// SNN (surrogate-gradient LIF) forward, restructured as 5 WMMA GEMMs + 5 scans
// Shapes: B=128, T=500, layers 784 -> 256 -> 128 -> 64 -> 32 -> 10
// ---------------------------------------------------------------------------

typedef __attribute__((ext_vector_type(16))) _Float16 v16h;
typedef __attribute__((ext_vector_type(8)))  float    v8f;
typedef int v4i_gcc __attribute__((vector_size(16)));   // matches builtin proto

#define BQ 128           // batch
#define TQ 500           // time steps
#define MTOT (TQ * BQ)   // 64000 rows in the flattened (t,b) GEMM

// -------- CDNA5 async global->LDS copy (ASYNCcnt path), with fallback -------
#if defined(__has_builtin)
#if __has_builtin(__builtin_amdgcn_global_load_async_to_lds_b128) && \
    __has_builtin(__builtin_amdgcn_s_wait_asynccnt)
#define USE_ASYNC_LDS 1
#endif
#endif
#ifndef USE_ASYNC_LDS
#define USE_ASYNC_LDS 0
#endif

#if USE_ASYNC_LDS
typedef __attribute__((address_space(1))) v4i_gcc* gptr_v4i;  // global int4*
typedef __attribute__((address_space(3))) v4i_gcc* lptr_v4i;  // LDS int4*

__device__ __forceinline__ void cp_async_b128(void* lds, const void* g) {
  __builtin_amdgcn_global_load_async_to_lds_b128(
      (gptr_v4i)(uintptr_t)g,
      (lptr_v4i)(unsigned)(uintptr_t)lds,   // low 32 bits = LDS offset
      0, 0);
}
#endif

__device__ __forceinline__ v8f wmma_f32_16x16x32_f16(v16h a, v16h b, v8f c) {
  return __builtin_amdgcn_wmma_f32_16x16x32_f16(
      /*neg_a=*/false, a, /*neg_b=*/false, b,
      /*c_mod=*/(short)0, c, /*reuse_a=*/false, /*reuse_b=*/false);
}

// ---- fragment gathers from LDS, following CDNA5 ISA 16-bit layouts ---------
// A (16x32 MxK, row-major LDS tile [128][32]): lane L holds M=L%16,
// K in {base..base+7} U {16+base..16+base+7}, base=(L<16)?0:8; pairs -> b32.
__device__ __forceinline__ v16h load_a_frag(const _Float16* As, int rowBase, int lane) {
  union { unsigned int u[8]; v16h h; } f;
  const unsigned int* row =
      (const unsigned int*)(As + (rowBase + (lane & 15)) * 32);
  const int base = (lane < 16) ? 0 : 4;   // dword index of first k-pair
#pragma unroll
  for (int j = 0; j < 4; ++j) f.u[j]     = row[base + j];
#pragma unroll
  for (int j = 0; j < 4; ++j) f.u[4 + j] = row[8 + base + j];
  return f.h;
}

// B (32x16 KxN): staged in LDS as [n][k] (64x32) so lane L (column N=L%16,
// K in {base..base+7} U {16+base..16+base+7}, base=(L<16)?0:8) reads k-pairs.
__device__ __forceinline__ v16h load_b_frag(const _Float16* Bs, int colBase, int lane) {
  union { unsigned int u[8]; v16h h; } f;
  const unsigned int* col =
      (const unsigned int*)(Bs + (colBase + (lane & 15)) * 32);
  const int base = (lane < 16) ? 0 : 4;
#pragma unroll
  for (int j = 0; j < 4; ++j) f.u[j]     = col[base + j];
#pragma unroll
  for (int j = 0; j < 4; ++j) f.u[4 + j] = col[8 + base + j];
  return f.h;
}

// ---- relu(W) -> f16, stored TRANSPOSED+PADDED: Wt[n][k], n<Npad, k<Kpad ----
__global__ void snn_prep_w(const float* __restrict__ w, _Float16* __restrict__ wt,
                           int K, int Kpad, int N, int Npad) {
  int idx = blockIdx.x * 256 + threadIdx.x;
  if (idx >= Npad * Kpad) return;
  int n = idx / Kpad, k = idx - n * Kpad;
  float v = (n < N && k < K) ? w[(size_t)k * N + n] : 0.f;
  wt[idx] = (_Float16)(v > 0.f ? v : 0.f);
}

// ---- (B, 784, T) f32 -> X0[(t*B+b)*800 + k] f16, k>=784 zeroed -------------
__global__ void snn_transpose_in(const float* __restrict__ in, _Float16* __restrict__ X0) {
  __shared__ float tile[32][33];
  const int t0 = blockIdx.x * 32, k0 = blockIdx.y * 32, b = blockIdx.z;
  for (int idx = threadIdx.x; idx < 1024; idx += 256) {
    int tt = idx & 31, kk = idx >> 5;
    int k = k0 + kk, t = t0 + tt;
    float v = 0.f;
    if (k < 784 && t < TQ) v = in[((size_t)b * 784 + k) * TQ + t];  // coalesced in t
    tile[kk][tt] = v;
  }
  __syncthreads();
  for (int idx = threadIdx.x; idx < 1024; idx += 256) {
    int kk = idx & 31, tt = idx >> 5;
    int t = t0 + tt;
    if (t < TQ)
      X0[((size_t)t * BQ + b) * 800 + k0 + kk] = (_Float16)tile[kk][tt]; // coalesced in k
  }
}

// ---- tiled WMMA GEMM: C[MTOT x N] = A[MTOT x Kpad](f16) @ Wt^T -------------
// Wt is [Npad][Kpad] f16 (N-major). WG tile 128x64, 8 waves of 32x32 each
// (2x2 WMMA frags). Double-buffered LDS, async global->LDS copies.
__global__ void __launch_bounds__(256)
snn_gemm(const _Float16* __restrict__ A, const _Float16* __restrict__ Wt,
         float* __restrict__ C, int N, int K, int lda, int ldwt, int ldc) {
  __shared__ _Float16 As[2][128 * 32];   // [buf][row][k]   8KB each
  __shared__ _Float16 Bs[2][64 * 32];    // [buf][n][k]     4KB each

  const int tid  = threadIdx.x;
  const int lane = tid & 31;
  const int wid  = tid >> 5;
  const int m0   = blockIdx.x * 128;
  const int n0   = blockIdx.y * 64;
  const int mbase = (wid & 3) * 32;
  const int nbase = (wid >> 2) * 32;

  // per-thread tile-copy addresses: 1x 16B chunk of B, 2x 16B chunks of A
  const int rA  = tid >> 2;          // 0..63
  const int koA = (tid & 3) * 8;     // half offset within a 32-half row
  const char* gA0 = (const char*)(A  + (size_t)(m0 + rA)      * lda  + koA);
  const char* gA1 = (const char*)(A  + (size_t)(m0 + 64 + rA) * lda  + koA);
  const char* gB  = (const char*)(Wt + (size_t)(n0 + rA)      * ldwt + koA);
  _Float16* lA0 = &As[0][rA * 32 + koA];
  _Float16* lA1 = &As[0][(64 + rA) * 32 + koA];
  _Float16* lB  = &Bs[0][rA * 32 + koA];
  const int STRA = 128 * 32;   // element stride between A buffers
  const int STRB = 64 * 32;    // element stride between B buffers

#if USE_ASYNC_LDS
#define TILE_COPY(bsel)                                      \
  do {                                                       \
    cp_async_b128(lA0 + (bsel) * STRA, gA0);                 \
    cp_async_b128(lA1 + (bsel) * STRA, gA1);                 \
    cp_async_b128(lB  + (bsel) * STRB, gB);                  \
    gA0 += 64; gA1 += 64; gB += 64;                          \
  } while (0)
#else
#define TILE_COPY(bsel)                                      \
  do {                                                       \
    uint4 ta = *(const uint4*)gA0;                           \
    uint4 tb = *(const uint4*)gA1;                           \
    uint4 tc = *(const uint4*)gB;                            \
    *(uint4*)(lA0 + (bsel) * STRA) = ta;                     \
    *(uint4*)(lA1 + (bsel) * STRA) = tb;                     \
    *(uint4*)(lB  + (bsel) * STRB) = tc;                     \
    gA0 += 64; gA1 += 64; gB += 64;                          \
  } while (0)
#endif

  v8f c00 = {}, c01 = {}, c10 = {}, c11 = {};

  const int KT = K >> 5;
  TILE_COPY(0);                       // prologue: tile 0 -> buffer 0

  for (int kt = 0; kt < KT; ++kt) {
    const int cur = kt & 1;
    if (kt + 1 < KT) TILE_COPY(cur ^ 1);   // prefetch next tile into other buf
#if USE_ASYNC_LDS
    if (kt + 1 < KT) __builtin_amdgcn_s_wait_asynccnt(3);  // cur tile landed
    else             __builtin_amdgcn_s_wait_asynccnt(0);
#endif
    __syncthreads();

    const _Float16* Ac = As[cur];
    const _Float16* Bc = Bs[cur];
    v16h a0 = load_a_frag(Ac, mbase,      lane);
    v16h a1 = load_a_frag(Ac, mbase + 16, lane);
    v16h b0 = load_b_frag(Bc, nbase,      lane);
    v16h b1 = load_b_frag(Bc, nbase + 16, lane);

    c00 = wmma_f32_16x16x32_f16(a0, b0, c00);
    c01 = wmma_f32_16x16x32_f16(a0, b1, c01);
    c10 = wmma_f32_16x16x32_f16(a1, b0, c10);
    c11 = wmma_f32_16x16x32_f16(a1, b1, c11);
    __syncthreads();                  // next iter overwrites buf `cur`
  }
#undef TILE_COPY

  // D layout: lane holds column N=lane%16, rows M=(lane<16?0:8)+r in VGPR r
  const int nn = lane & 15;
  const int mo = (lane >> 4) << 3;
#pragma unroll
  for (int i = 0; i < 2; ++i)
#pragma unroll
    for (int j = 0; j < 2; ++j) {
      int n = n0 + nbase + 16 * j + nn;
      if (n < N) {
        v8f acc = (i == 0) ? (j == 0 ? c00 : c01) : (j == 0 ? c10 : c11);
        float* cp = C + (size_t)(m0 + mbase + 16 * i + mo) * ldc + n;
#pragma unroll
        for (int r = 0; r < 8; ++r) cp[(size_t)r * ldc] = acc[r];
      }
    }
}

// ---- LIF scan: sequential in t, parallel over (b,h) ------------------------
// v = 0.5*(v + cur); s = sigmoid(5*(v-1)); v *= (1-s)   [V_REST=V_RESET=0]
__global__ void snn_lif_scan(const float* __restrict__ Cur, int ldc, int H,
                             float* __restrict__ outS,        // (B,H,T) f32
                             _Float16* __restrict__ Xnext,    // shifted (T*B,H) f16 or null
                             float* __restrict__ fr) {        // (B,H) or null
  int idx = blockIdx.x * 256 + threadIdx.x;
  if (idx >= BQ * H) return;
  int b = idx / H, h = idx - b * H;

  if (Xnext) Xnext[(size_t)b * H + h] = (_Float16)0.f;  // t=0 input row = zeros

  float v = 0.f, sum = 0.f;
  float* o = outS + ((size_t)b * H + h) * TQ;
  for (int t = 0; t < TQ; ++t) {
    float c = Cur[((size_t)t * BQ + b) * ldc + h];      // coalesced across h
    v = 0.5f * (v + c);
    float s = 1.f / (1.f + __expf(-5.f * (v - 1.f)));
    v = v * (1.f - s);
    o[t] = s;
    sum += s;
    if (Xnext && t < TQ - 1)
      Xnext[((size_t)(t + 1) * BQ + b) * H + h] = (_Float16)s;
  }
  if (fr) fr[idx] = sum * (1.f / (float)TQ);
}

// ---------------------------------------------------------------------------
extern "C" void kernel_launch(void* const* d_in, const int* in_sizes, int n_in,
                              void* d_out, int out_size, void* d_ws, size_t ws_size,
                              hipStream_t stream) {
  const float* in_sp = (const float*)d_in[0];
  const float* w0 = (const float*)d_in[1];
  const float* w1 = (const float*)d_in[2];
  const float* w2 = (const float*)d_in[3];
  const float* w3 = (const float*)d_in[4];
  const float* w4 = (const float*)d_in[5];
  float* out = (float*)d_out;
  char* ws = (char*)d_ws;

  // output section offsets (floats)
  const size_t OFF_H0 = 0;
  const size_t OFF_H1 = (size_t)BQ * 256 * TQ;             // 16,384,000
  const size_t OFF_H2 = OFF_H1 + (size_t)BQ * 128 * TQ;    // 24,576,000
  const size_t OFF_H3 = OFF_H2 + (size_t)BQ * 64 * TQ;     // 28,672,000
  const size_t OFF_O  = OFF_H3 + (size_t)BQ * 32 * TQ;     // 30,720,000
  const size_t OFF_FR = OFF_O + (size_t)BQ * 10 * TQ;      // 31,360,000

  // scratch layout (256B aligned). Weights stored [Npad][Kpad] f16.
  const size_t O_W0 = 0;                                   // 256 x 800
  const size_t O_W1 = O_W0 + 409600;                       // 128 x 256
  const size_t O_W2 = O_W1 + 65536;                        // 64  x 128
  const size_t O_W3 = O_W2 + 16384;                        // 64  x 64
  const size_t O_W4 = O_W3 + 8192;                         // 64  x 32
  const size_t O_A  = O_W4 + 4096;                         // 102.4 MB : X0 / X2 / X4
  const size_t O_B  = O_A + (size_t)MTOT * 800 * 2;        // 65.5 MB : C0 / C2 / C4
  const size_t O_C  = O_B + (size_t)MTOT * 256 * 4;        // 32.8 MB : X1 / X3
  const size_t O_D  = O_C + (size_t)MTOT * 256 * 2;        // 32.8 MB : C1 / C3

  _Float16* W0t = (_Float16*)(ws + O_W0);
  _Float16* W1t = (_Float16*)(ws + O_W1);
  _Float16* W2t = (_Float16*)(ws + O_W2);
  _Float16* W3t = (_Float16*)(ws + O_W3);
  _Float16* W4t = (_Float16*)(ws + O_W4);
  _Float16* X0  = (_Float16*)(ws + O_A);
  _Float16* X1  = (_Float16*)(ws + O_C);
  _Float16* X2  = (_Float16*)(ws + O_A);
  _Float16* X3  = (_Float16*)(ws + O_C);
  _Float16* X4  = (_Float16*)(ws + O_A);
  float* C0 = (float*)(ws + O_B);
  float* C1 = (float*)(ws + O_D);
  float* C2 = (float*)(ws + O_B);
  float* C3 = (float*)(ws + O_D);
  float* C4 = (float*)(ws + O_B);

  // 1) weights: relu -> f16, transposed to [Npad][Kpad]
  snn_prep_w<<<(256 * 800 + 255) / 256, 256, 0, stream>>>(w0, W0t, 784, 800, 256, 256);
  snn_prep_w<<<(128 * 256 + 255) / 256, 256, 0, stream>>>(w1, W1t, 256, 256, 128, 128);
  snn_prep_w<<<(64  * 128 + 255) / 256, 256, 0, stream>>>(w2, W2t, 128, 128, 64, 64);
  snn_prep_w<<<(64  * 64  + 255) / 256, 256, 0, stream>>>(w3, W3t, 64,  64,  32, 64);
  snn_prep_w<<<(64  * 32  + 255) / 256, 256, 0, stream>>>(w4, W4t, 32,  32,  10, 64);

  // 2) input (B,784,T) -> X0 (T*B, 800) f16
  snn_transpose_in<<<dim3(16, 25, BQ), 256, 0, stream>>>(in_sp, X0);

  // 3) layer pipeline: GEMM (WMMA) + LIF scan, spikes time-shifted for next layer
  snn_gemm<<<dim3(MTOT / 128, 4), 256, 0, stream>>>(X0, W0t, C0, 256, 800, 800, 800, 256);
  snn_lif_scan<<<(BQ * 256 + 255) / 256, 256, 0, stream>>>(C0, 256, 256, out + OFF_H0, X1, nullptr);

  snn_gemm<<<dim3(MTOT / 128, 2), 256, 0, stream>>>(X1, W1t, C1, 128, 256, 256, 256, 128);
  snn_lif_scan<<<(BQ * 128 + 255) / 256, 256, 0, stream>>>(C1, 128, 128, out + OFF_H1, X2, nullptr);

  snn_gemm<<<dim3(MTOT / 128, 1), 256, 0, stream>>>(X2, W2t, C2, 64, 128, 128, 128, 64);
  snn_lif_scan<<<(BQ * 64 + 255) / 256, 256, 0, stream>>>(C2, 64, 64, out + OFF_H2, X3, nullptr);

  snn_gemm<<<dim3(MTOT / 128, 1), 256, 0, stream>>>(X3, W3t, C3, 32, 64, 64, 64, 32);
  snn_lif_scan<<<(BQ * 32 + 255) / 256, 256, 0, stream>>>(C3, 32, 32, out + OFF_H3, X4, nullptr);

  snn_gemm<<<dim3(MTOT / 128, 1), 256, 0, stream>>>(X4, W4t, C4, 10, 32, 32, 32, 10);
  snn_lif_scan<<<(BQ * 10 + 255) / 256, 256, 0, stream>>>(C4, 10, 10, out + OFF_O, nullptr, out + OFF_FR);

  (void)in_sizes; (void)n_in; (void)out_size; (void)ws_size;
}
